// PI0Pytorch_51582557225649
// MI455X (gfx1250) — matrix operand order repaired and checked by reading, not compiled, not run on hardware
//
#include <hip/hip_runtime.h>
#include <hip/hip_bf16.h>
#include <math.h>
#include <stdint.h>

// ---------------------------------------------------------------------------
// PI0 action-expert slice on MI455X (gfx1250, wave32, WMMA).
//
// Only the 50 action-token query rows feed the output => Q/scores/PV/Wo run on
// a 64-row padded slice per batch; K/V projections over all 867 tokens
// dominate (~116 of ~135 GFLOP). All GEMMs use v_wmma_f32_16x16x32_bf16 with
// fp32 accumulate. fp32->bf16 conversion happens ONCE per tensor (weights are
// also pre-transposed to [N][K]). The unguarded GEMM stages tiles with
// global_load_async_to_lds_b128 (ASYNCcnt) into double-buffered LDS: one
// workgroup barrier per K-step, no VGPR transit for operand data.
// ---------------------------------------------------------------------------

typedef unsigned short u16;
typedef __attribute__((ext_vector_type(8)))  float  v8f_t;
typedef __attribute__((ext_vector_type(16))) __bf16 v16bf_t;
typedef __attribute__((ext_vector_type(8)))  u16    u16x8_t;

#define BM 64
#define BN 128
#define BK 32
#define ASTR 40   // halves per LDS row; 80B stride keeps 16B alignment
#define BSTR 40

#define SPAD 896          // 867 tokens padded to 28*32
#define DTOK 2048
#define NB 8
#define NH 8
#define HD 256
#define HACT 50
#define PREF 816
#define SVALID 867        // P + 1 + H_ACT
#define MTOK (NB * SPAD)  // 7168

__device__ __forceinline__ u16 f2bf(float x) {
  unsigned int u = __float_as_uint(x);
  unsigned int r = ((u >> 16) & 1u) + 0x7FFFu;   // round-to-nearest-even
  return (u16)((u + r) >> 16);
}

// ---------------------------------------------------------------------------
// Batched bf16 WMMA GEMM: C = act( scale * A*B + bias )
//   A: bf16 [M x K] (lda, halves), B: bf16 [N x K] (ldb)  -- B pre-transposed.
//   outMode: 0 = fp32 [M x N], 1 = bf16 [M x N], 2 = bf16 transposed [N x M]
//   z = (zb, zh): pointer offsets zb*OffB + zh*OffH (per-(batch,head) GEMMs).
// GUARD=false requires M%64==0 && N%128==0 and uses the async-to-LDS path.
// ---------------------------------------------------------------------------
template <bool GUARD>
__global__ __launch_bounds__(256) void gemm_bf16_wmma(
    const u16* __restrict__ Ag, const u16* __restrict__ Bg,
    const float* __restrict__ bias, void* __restrict__ Cg,
    int M, int N, int K, int lda, int ldb, int ldc,
    long long aOffB, long long aOffH,
    long long bOffB, long long bOffH,
    long long cOffB, long long cOffH,
    int hcount, int actSilu, int outMode, float scale)
{
  __shared__ __align__(16) u16 As[2][BM * ASTR];   // GUARD path uses [0] only
  __shared__ __align__(16) u16 Bs[2][BN * BSTR];

  const int z  = blockIdx.z;
  const int zb = z / hcount;
  const int zh = z - zb * hcount;
  const u16* A = Ag + zb * aOffB + zh * aOffH;
  const u16* B = Bg + zb * bOffB + zh * bOffH;

  const int tid     = threadIdx.x;
  const int rowBase = blockIdx.y * BM;
  const int colBase = blockIdx.x * BN;

  // staging assignments: A = 64 rows x 4 chunks of 8 halves; B = 128 x 4.
  const int ar = tid >> 2;
  const int ak = (tid & 3) << 3;
  const int c1 = tid + 256;
  const int n0 = tid >> 2, k0 = (tid & 3) << 3;
  const int n1 = c1 >> 2,  k1 = (c1 & 3) << 3;

  const int wid  = tid >> 5;
  const int lane = tid & 31;
  const int wr   = wid >> 1;     // 4 row groups x 16 rows
  const int wc   = wid & 1;      // 2 col groups x 64 cols
  const int l15  = lane & 15;
  const int hi   = lane >> 4;

  v8f_t acc[4];
  for (int i = 0; i < 4; ++i)
    for (int j = 0; j < 8; ++j) acc[i][j] = 0.0f;

  // fragment loads + 4 WMMAs for one staged K-tile
  auto compute_tile = [&](const u16* as, const u16* bs) {
    union { v16bf_t v; u16x8_t h[2]; } af;
    const int am = wr * 16 + l15;
    af.h[0] = *(const u16x8_t*)&as[am * ASTR + hi * 8];
    af.h[1] = *(const u16x8_t*)&as[am * ASTR + 16 + hi * 8];
#pragma unroll
    for (int bi = 0; bi < 4; ++bi) {
      union { v16bf_t v; u16x8_t h[2]; } bfb;
      const int bnn = wc * 64 + bi * 16 + l15;
      bfb.h[0] = *(const u16x8_t*)&bs[bnn * BSTR + hi * 8];
      bfb.h[1] = *(const u16x8_t*)&bs[bnn * BSTR + 16 + hi * 8];
      acc[bi] = __builtin_amdgcn_wmma_f32_16x16x32_bf16(
          false, af.v, false, bfb.v, (short)0, acc[bi], false, false);
    }
  };

  if constexpr (!GUARD) {
    // ---- async global->LDS staging, double buffered ----
    const u16* pA  = A + (size_t)(rowBase + ar) * lda + ak;
    const u16* pB0 = B + (size_t)(colBase + n0) * ldb + k0;
    const u16* pB1 = B + (size_t)(colBase + n1) * ldb + k1;

    auto issue = [&](int kt, int buf) {
      unsigned la  = (unsigned)(uintptr_t)&As[buf][ar * ASTR + ak];
      unsigned lb0 = (unsigned)(uintptr_t)&Bs[buf][n0 * BSTR + k0];
      unsigned lb1 = (unsigned)(uintptr_t)&Bs[buf][n1 * BSTR + k1];
      asm volatile("global_load_async_to_lds_b128 %0, %1, off"
                   :: "v"(la),
                      "v"((unsigned long long)(uintptr_t)(pA + kt))
                   : "memory");
      asm volatile("global_load_async_to_lds_b128 %0, %1, off"
                   :: "v"(lb0),
                      "v"((unsigned long long)(uintptr_t)(pB0 + kt))
                   : "memory");
      asm volatile("global_load_async_to_lds_b128 %0, %1, off"
                   :: "v"(lb1),
                      "v"((unsigned long long)(uintptr_t)(pB1 + kt))
                   : "memory");
    };

    issue(0, 0);
    const int nIter = K >> 5;
    for (int it = 0; it < nIter; ++it) {
      asm volatile("s_wait_asynccnt 0x0" ::: "memory");
      __syncthreads();                 // tile `it` visible to all waves
      if (it + 1 < nIter) issue((it + 1) << 5, (it + 1) & 1);
      const int buf = it & 1;
      compute_tile(As[buf], Bs[buf]);
      // no trailing barrier: each wave's ds_loads complete before it signals
      // the next iteration's barrier, after which buffer buf is overwritten.
    }
  } else {
    // ---- guarded register-prefetch staging (partial tiles; zero fill) ----
    auto loadA = [&](int kt) -> u16x8_t {
      u16x8_t v = (u16x8_t)(u16)0;
      int gr = rowBase + ar;
      if (gr < M) v = *(const u16x8_t*)(A + (size_t)gr * lda + kt + ak);
      return v;
    };
    auto loadB = [&](int kt, int n, int kk) -> u16x8_t {
      u16x8_t v = (u16x8_t)(u16)0;
      int gn = colBase + n;
      if (gn < N) v = *(const u16x8_t*)(B + (size_t)gn * ldb + kt + kk);
      return v;
    };
    u16x8_t ra  = loadA(0);
    u16x8_t rb0 = loadB(0, n0, k0);
    u16x8_t rb1 = loadB(0, n1, k1);
    for (int kt = 0; kt < K; kt += BK) {
      *(u16x8_t*)&As[0][ar * ASTR + ak] = ra;
      *(u16x8_t*)&Bs[0][n0 * BSTR + k0] = rb0;
      *(u16x8_t*)&Bs[0][n1 * BSTR + k1] = rb1;
      __syncthreads();
      int ktn = kt + BK;
      if (ktn < K) {
        ra  = loadA(ktn);
        rb0 = loadB(ktn, n0, k0);
        rb1 = loadB(ktn, n1, k1);
      }
      compute_tile(As[0], Bs[0]);
      __syncthreads();
    }
  }

  // ---- epilogue: D layout: vgpr r -> M = wr*16 + hi*8 + r ----
#pragma unroll
  for (int bi = 0; bi < 4; ++bi) {
    int gn = colBase + wc * 64 + bi * 16 + l15;
    if (GUARD && gn >= N) continue;
    float bv = bias ? bias[gn] : 0.0f;
    const int gm0 = rowBase + wr * 16 + hi * 8;

    if (outMode == 2 && !GUARD) {
      // transposed bf16 store: Ct[gn][gm0 .. gm0+7] contiguous -> one b128
      union { u16x8_t v; u16 s[8]; } o;
#pragma unroll
      for (int r = 0; r < 8; ++r) {
        float v = acc[bi][r] * scale + bv;
        if (actSilu) v = v / (1.0f + __expf(-v));
        o.s[r] = f2bf(v);
      }
      *(u16x8_t*)((u16*)Cg + zb * cOffB + zh * cOffH +
                  (size_t)gn * ldc + gm0) = o.v;
    } else {
#pragma unroll
      for (int r = 0; r < 8; ++r) {
        int gm = gm0 + r;
        if (GUARD && gm >= M) continue;
        float v = acc[bi][r] * scale + bv;
        if (actSilu) v = v / (1.0f + __expf(-v));
        if (outMode == 0)
          ((float*)Cg)[zb * cOffB + zh * cOffH + (size_t)gm * ldc + gn] = v;
        else if (outMode == 1)
          ((u16*)Cg)[zb * cOffB + zh * cOffH + (size_t)gm * ldc + gn] = f2bf(v);
        else
          ((u16*)Cg)[zb * cOffB + zh * cOffH + (size_t)gn * ldc + gm] = f2bf(v);
      }
    }
  }
}

// ---------------------------------------------------------------------------
// One-time fp32 -> bf16 weight convert + transpose: Wt[n][k] = W[k][n]
// ---------------------------------------------------------------------------
__global__ __launch_bounds__(256) void wconv_t(const float* __restrict__ W,
                                               u16* __restrict__ Wt,
                                               int K, int N) {
  __shared__ u16 t[32][33];
  int k0 = blockIdx.y * 32, n0 = blockIdx.x * 32;
  int tid = threadIdx.x;
#pragma unroll
  for (int i = 0; i < 4; ++i) {
    int l = tid + i * 256;
    int k = l >> 5, n = l & 31;
    t[k][n] = f2bf(W[(long long)(k0 + k) * N + n0 + n]);
  }
  __syncthreads();
#pragma unroll
  for (int i = 0; i < 4; ++i) {
    int l = tid + i * 256;
    int n = l >> 5, k = l & 31;
    Wt[(long long)(n0 + n) * K + k0 + k] = t[k][n];
  }
}

// ---------------------------------------------------------------------------
// Small producer kernels (all emit bf16)
// ---------------------------------------------------------------------------

__global__ void init_tokens(const float4* __restrict__ prefix,
                            uint2* __restrict__ tokens, int tot4) {
  int i = blockIdx.x * blockDim.x + threadIdx.x;
  if (i >= tot4) return;
  int row = i >> 9;            // 512 x float4 per 2048-wide row
  int d4  = i & 511;
  int b = row / SPAD, p = row % SPAD;
  uint2 o = make_uint2(0u, 0u);
  if (p < PREF) {
    float4 v = prefix[((long long)(b * PREF + p)) * 512 + d4];
    o.x = (unsigned)f2bf(v.x) | ((unsigned)f2bf(v.y) << 16);
    o.y = (unsigned)f2bf(v.z) | ((unsigned)f2bf(v.w) << 16);
  }
  tokens[i] = o;
}

__global__ void state_row(const float* __restrict__ st,
                          const float* __restrict__ W,
                          const float* __restrict__ bias,
                          u16* __restrict__ tokens) {
  int i = blockIdx.x * blockDim.x + threadIdx.x;
  if (i >= NB * DTOK) return;
  int b = i >> 11, d = i & 2047;
  float acc = bias[d];
#pragma unroll
  for (int k = 0; k < 32; ++k) acc += st[b * 32 + k] * W[k * DTOK + d];
  tokens[((long long)(b * SPAD + PREF)) * DTOK + d] = f2bf(acc);
}

__global__ void build_xcat(const float* __restrict__ na,
                           const float* __restrict__ W,
                           const float* __restrict__ bias,
                           const float* __restrict__ timev,
                           u16* __restrict__ xcat) {
  int i = blockIdx.x * blockDim.x + threadIdx.x;
  if (i >= NB * HACT * 4096) return;
  int row = i >> 12;
  int d   = i & 4095;
  int b = row / HACT, h = row - b * HACT;
  float v;
  if (d < DTOK) {
    v = bias[d];
#pragma unroll
    for (int k = 0; k < 32; ++k)
      v += na[(b * HACT + h) * 32 + k] * W[k * DTOK + d];
  } else {
    int d2 = d - DTOK;
    int j = d2 & 1023;
    float f = (float)j * (1.0f / 1023.0f);
    float s = 1570.7963267948966f * __expf(-f * 6.907755279f);
    float t = timev[b];
    v = (d2 >= 1024) ? __cosf(s * t) : __sinf(s * t);
  }
  xcat[i] = f2bf(v);
}

__global__ void scatter_act(const u16* __restrict__ acttok,
                            u16* __restrict__ tokens) {
  int i = blockIdx.x * blockDim.x + threadIdx.x;
  if (i >= NB * HACT * DTOK) return;
  int row = i >> 11, d = i & 2047;
  int b = row / HACT, h = row - b * HACT;
  tokens[((long long)(b * SPAD + PREF + 1 + h)) * DTOK + d] = acttok[i];
}

__global__ void extract_q(const u16* __restrict__ tokens,
                          u16* __restrict__ qsrc) {
  int i = blockIdx.x * blockDim.x + threadIdx.x;
  if (i >= NB * 64 * DTOK) return;
  int row = i >> 11, d = i & 2047;
  int b = row >> 6, r = row & 63;
  qsrc[i] = tokens[((long long)(b * SPAD + PREF + r)) * DTOK + d];
}

__global__ void softmax_mask(float* __restrict__ scores,
                             u16* __restrict__ probs,
                             const unsigned char* __restrict__ pad) {
  int r = blockIdx.x, h = blockIdx.y, b = blockIdx.z;
  long long base = ((long long)((b * NH + h) * 64 + r)) * SPAD;
  float* row = scores + base;
  u16*   prw = probs + base;
  __shared__ float red[256];
  int tid = threadIdx.x;

  float m = -3.4e38f;
  for (int k = tid; k < SPAD; k += 256) {
    bool valid = (k < SVALID) && (k >= PREF || pad[b * PREF + k] != 0);
    if (valid) m = fmaxf(m, row[k]);
  }
  red[tid] = m;
  __syncthreads();
  for (int s = 128; s > 0; s >>= 1) {
    if (tid < s) red[tid] = fmaxf(red[tid], red[tid + s]);
    __syncthreads();
  }
  float mx = red[0];
  __syncthreads();

  float sum = 0.0f;
  for (int k = tid; k < SPAD; k += 256) {
    bool valid = (k < SVALID) && (k >= PREF || pad[b * PREF + k] != 0);
    float p = valid ? __expf(row[k] - mx) : 0.0f;
    row[k] = p;
    sum += p;
  }
  red[tid] = sum;
  __syncthreads();
  for (int s = 128; s > 0; s >>= 1) {
    if (tid < s) red[tid] += red[tid + s];
    __syncthreads();
  }
  float inv = (red[0] > 0.0f) ? (1.0f / red[0]) : 0.0f;
  for (int k = tid; k < SPAD; k += 256) prw[k] = f2bf(row[k] * inv);
}

__global__ void vt_out(const float* __restrict__ proj,
                       const float* __restrict__ W,
                       const float* __restrict__ bias,
                       float* __restrict__ out) {
  int i = blockIdx.x * blockDim.x + threadIdx.x;
  if (i >= NB * HACT * 32) return;
  int a = i & 31;
  int row = i >> 5;
  int b = row / HACT, h = row - b * HACT;
  const float* p = proj + ((long long)(b * 64 + 1 + h)) * DTOK;
  float acc = bias[a];
  for (int k = 0; k < DTOK; ++k) acc += p[k] * W[k * 32 + a];
  out[i] = acc;
}

// ---------------------------------------------------------------------------
static void launch_gemm(hipStream_t s, const u16* A, const u16* B,
                        const float* bias, void* C,
                        int M, int N, int K, int lda, int ldb, int ldc,
                        long long aOffB, long long aOffH,
                        long long bOffB, long long bOffH,
                        long long cOffB, long long cOffH,
                        int hcount, int Z, int actSilu, int outMode,
                        float scale) {
  dim3 g((unsigned)((N + BN - 1) / BN), (unsigned)((M + BM - 1) / BM),
         (unsigned)Z);
  bool guard = (M & (BM - 1)) || (N & (BN - 1));
  if (guard)
    gemm_bf16_wmma<true><<<g, 256, 0, s>>>(A, B, bias, C, M, N, K, lda, ldb,
        ldc, aOffB, aOffH, bOffB, bOffH, cOffB, cOffH, hcount, actSilu,
        outMode, scale);
  else
    gemm_bf16_wmma<false><<<g, 256, 0, s>>>(A, B, bias, C, M, N, K, lda, ldb,
        ldc, aOffB, aOffH, bOffB, bOffH, cOffB, cOffH, hcount, actSilu,
        outMode, scale);
}

extern "C" void kernel_launch(void* const* d_in, const int* in_sizes, int n_in,
                              void* d_out, int out_size, void* d_ws,
                              size_t ws_size, hipStream_t stream) {
  (void)in_sizes; (void)n_in; (void)out_size; (void)ws_size;
  const float* prefix = (const float*)d_in[0];
  const float* state  = (const float*)d_in[1];
  const float* nacts  = (const float*)d_in[2];
  const float* timev  = (const float*)d_in[3];
  const float* Wq     = (const float*)d_in[4];
  const float* Wk     = (const float*)d_in[5];
  const float* Wv     = (const float*)d_in[6];
  const float* Wo     = (const float*)d_in[7];
  const float* spw    = (const float*)d_in[8];
  const float* spb    = (const float*)d_in[9];
  const float* aiw    = (const float*)d_in[10];
  const float* aib    = (const float*)d_in[11];
  const float* amiw   = (const float*)d_in[12];
  const float* amib   = (const float*)d_in[13];
  const float* amow   = (const float*)d_in[14];
  const float* amob   = (const float*)d_in[15];
  const float* aow    = (const float*)d_in[16];
  const float* aob    = (const float*)d_in[17];
  const unsigned char* ppad = (const unsigned char*)d_in[18];
  // d_in[19] (att_masks) unused: action queries attend every valid key.

  float* out = (float*)d_out;

  // ---- workspace carve-up (all offsets 16B aligned) ----
  char* p = (char*)d_ws;
  auto carve = [&](size_t bytes) { char* r = p; p += bytes; return r; };
  u16*   tokens = (u16*)carve((size_t)MTOK * DTOK * 2);
  u16*   kbuf   = (u16*)carve((size_t)MTOK * DTOK * 2);
  u16*   vbufT  = (u16*)carve((size_t)DTOK * MTOK * 2);
  u16*   qsrc   = (u16*)carve((size_t)512 * DTOK * 2);
  u16*   qbuf   = (u16*)carve((size_t)512 * DTOK * 2);
  u16*   xcat   = (u16*)carve((size_t)400 * 4096 * 2);
  u16*   hid    = (u16*)carve((size_t)400 * DTOK * 2);
  u16*   acttok = (u16*)carve((size_t)400 * DTOK * 2);
  float* scores = (float*)carve((size_t)NB * NH * 64 * SPAD * 4);
  u16*   probs  = (u16*)carve((size_t)NB * NH * 64 * SPAD * 2);
  u16*   attnc  = (u16*)carve((size_t)512 * DTOK * 2);
  float* projb  = (float*)carve((size_t)512 * DTOK * 4);
  u16*   WqT    = (u16*)carve((size_t)DTOK * DTOK * 2);
  u16*   WkT    = (u16*)carve((size_t)DTOK * DTOK * 2);
  u16*   WvT    = (u16*)carve((size_t)DTOK * DTOK * 2);
  u16*   WoT    = (u16*)carve((size_t)DTOK * DTOK * 2);
  u16*   amiwT  = (u16*)carve((size_t)DTOK * 4096 * 2);
  u16*   amowT  = (u16*)carve((size_t)DTOK * DTOK * 2);

  // ---- one-time weight convert + transpose (fp32 -> bf16 [N][K]) ----
  {
    dim3 g(DTOK / 32, DTOK / 32);
    wconv_t<<<g, 256, 0, stream>>>(Wq, WqT, DTOK, DTOK);
    wconv_t<<<g, 256, 0, stream>>>(Wk, WkT, DTOK, DTOK);
    wconv_t<<<g, 256, 0, stream>>>(Wv, WvT, DTOK, DTOK);
    wconv_t<<<g, 256, 0, stream>>>(Wo, WoT, DTOK, DTOK);
    wconv_t<<<g, 256, 0, stream>>>(amow, amowT, DTOK, DTOK);
    dim3 g2(DTOK / 32, 4096 / 32);
    wconv_t<<<g2, 256, 0, stream>>>(amiw, amiwT, 4096, DTOK);
  }

  // ---- assemble padded bf16 token matrix ----
  {
    int tot4 = NB * SPAD * (DTOK / 4);
    init_tokens<<<(tot4 + 255) / 256, 256, 0, stream>>>(
        (const float4*)prefix, (uint2*)tokens, tot4);
  }
  state_row<<<(NB * DTOK + 255) / 256, 256, 0, stream>>>(state, spw, spb,
                                                         tokens);
  build_xcat<<<(NB * HACT * 4096 + 255) / 256, 256, 0, stream>>>(
      nacts, aiw, aib, timev, xcat);

  // ---- action-token MLP: silu(xcat @ atm_in) @ atm_out ----
  launch_gemm(stream, xcat, amiwT, amib, hid, 400, DTOK, 4096, 4096, 4096,
              DTOK, 0, 0, 0, 0, 0, 0, 1, 1, 1, 1, 1.0f);
  launch_gemm(stream, hid, amowT, amob, acttok, 400, DTOK, DTOK, DTOK, DTOK,
              DTOK, 0, 0, 0, 0, 0, 0, 1, 1, 0, 1, 1.0f);
  scatter_act<<<(NB * HACT * DTOK + 255) / 256, 256, 0, stream>>>(acttok,
                                                                  tokens);
  extract_q<<<(NB * 64 * DTOK + 255) / 256, 256, 0, stream>>>(tokens, qsrc);

  // ---- projections: K, V (V stored transposed), Q slice ----
  launch_gemm(stream, tokens, WkT, nullptr, kbuf, MTOK, DTOK, DTOK, DTOK, DTOK,
              DTOK, 0, 0, 0, 0, 0, 0, 1, 1, 0, 1, 1.0f);
  launch_gemm(stream, tokens, WvT, nullptr, vbufT, MTOK, DTOK, DTOK, DTOK,
              DTOK, MTOK /*ldc of V^T*/, 0, 0, 0, 0, 0, 0, 1, 1, 0, 2, 1.0f);
  launch_gemm(stream, qsrc, WqT, nullptr, qbuf, 512, DTOK, DTOK, DTOK, DTOK,
              DTOK, 0, 0, 0, 0, 0, 0, 1, 1, 0, 1, 1.0f);

  // ---- scores = (Q K^T)/16 per (b,h): M=64, N=896, K=256 ----
  launch_gemm(stream, qbuf, kbuf, nullptr, scores, 64, SPAD, HD, DTOK, DTOK,
              SPAD,
              64LL * DTOK, 256,                       // A: b,h offsets
              (long long)SPAD * DTOK, 256,            // B: kbuf b,h offsets
              (long long)NH * 64 * SPAD, 64LL * SPAD, // C: fp32 elements
              NH, NB * NH, 0, 0, 0.0625f);

  // ---- masked softmax -> bf16 probs ----
  {
    dim3 g(64, NH, NB);
    softmax_mask<<<g, 256, 0, stream>>>(scores, probs, ppad);
  }

  // ---- attn = probs @ V per (b,h): M=64, N=256, K=896 (B = V^T rows) ----
  launch_gemm(stream, probs, vbufT, nullptr, attnc, 64, HD, SPAD, SPAD,
              MTOK /*ldb of V^T*/, DTOK,
              (long long)NH * 64 * SPAD, 64LL * SPAD,   // A: probs offsets
              896LL /*col shift per b*/, 256LL * MTOK,  // B: V^T offsets
              64LL * DTOK, 256,                         // C: bf16 elements
              NH, NB * NH, 0, 1, 1.0f);

  // ---- output projection Wo on the 512-row slice -> fp32 ----
  launch_gemm(stream, attnc, WoT, nullptr, projb, 512, DTOK, DTOK, DTOK, DTOK,
              DTOK, 0, 0, 0, 0, 0, 0, 1, 1, 0, 0, 1.0f);

  // ---- decode velocities ----
  vt_out<<<(NB * HACT * 32 + 255) / 256, 256, 0, stream>>>(projb, aow, aob,
                                                           out);
}
